// MultiHeadAttention_27298812133654
// MI455X (gfx1250) — compile-verified, matrix operand order checked by reading, not run on hardware
//
#include <hip/hip_runtime.h>
#include <hip/hip_bf16.h>

typedef __bf16 bf16_t;
typedef __attribute__((ext_vector_type(16))) __bf16 v16bf;
typedef __attribute__((ext_vector_type(8)))  __bf16 v8bf;
typedef __attribute__((ext_vector_type(4)))  __bf16 v4bf;
typedef __attribute__((ext_vector_type(2)))  __bf16 v2bf;
typedef __attribute__((ext_vector_type(8)))  float  v8f;

#define BB  4
#define TT  2048
#define DD  1024
#define HH  16
#define DH  64

union FragU { v16bf v; v8bf h[2]; };

// Load a 16x32 bf16 WMMA A/B fragment from a row-major [rows][ld] bf16 matrix.
// A: row = base_row + lane%16 ; B: n = base_row + lane%16 with source in [N][K] layout.
// Per documented layout: elements 0..7 at K=k0+(lane/16)*8, elements 8..15 at K=k0+16+(lane/16)*8.
__device__ inline v16bf load_frag(const bf16_t* base, int row0, int k0, int ld) {
  const int lane = threadIdx.x & 31;
  const int sel  = lane & 15;
  const int hf   = lane >> 4;
  const bf16_t* p = base + (size_t)(row0 + sel) * ld + k0 + hf * 8;
  FragU f;
  f.h[0] = *reinterpret_cast<const v8bf*>(p);
  f.h[1] = *reinterpret_cast<const v8bf*>(p + 16);
  return f.v;
}

__device__ inline v8f wmma_bf16(v16bf a, v16bf b, v8f c) {
  // (neg_a, A, neg_b, B, c_mod, C, reuse_a, reuse_b)
  return __builtin_amdgcn_wmma_f32_16x16x32_bf16(false, a, false, b, (short)0, c, false, false);
}

// ---------------------------------------------------------------------------
// Kernel 1: fused QKV projection.  X[8192,1024]f32 x W[h][1024,64]f32 -> bf16
// Q,K stored [b,h,t,dh] row-major; V stored transposed [b,h,dh,t].
// 1/sqrt(DH)=0.125 folded into Q.
// grid (64, 48): x = 8192/128 row tiles, y = (mat 0..2)*16 + head
// ---------------------------------------------------------------------------
__global__ __launch_bounds__(256)
void mha_qkv_proj_kernel(const float* __restrict__ x,
                         const float* __restrict__ Wq,
                         const float* __restrict__ Wk,
                         const float* __restrict__ Wv,
                         bf16_t* __restrict__ Qb,
                         bf16_t* __restrict__ Kb,
                         bf16_t* __restrict__ Vt) {
  __shared__ bf16_t Xs[128 * 72];   // 128 rows x 64 K (+8 pad)
  __shared__ bf16_t Ws[64 * 72];    // [n=64][k=64] transposed weight tile
  const int tid  = threadIdx.x;
  const int lane = tid & 31, wid = tid >> 5;
  const int lanen = lane & 15, hf = lane >> 4;
  const int wm = wid >> 1, wn = wid & 1;          // 4x2 wave grid -> 128x64 tile
  const int r0  = blockIdx.x * 128;
  const int mh  = blockIdx.y;
  const int mat = mh >> 4;                        // 0=Q 1=K 2=V
  const int h   = mh & 15;
  const float* W = (mat == 0 ? Wq : (mat == 1 ? Wk : Wv)) + (size_t)h * DD * DH;

  v8f c[2][2] = {};

  for (int k0 = 0; k0 < DD; k0 += 64) {
    // Stage X tile (convert f32 -> bf16)
#pragma unroll
    for (int i = 0; i < 8; ++i) {
      int idx = i * 256 + tid;               // 2048 float4 loads
      int row = idx >> 4;
      int c4  = idx & 15;
      float4 xv = *reinterpret_cast<const float4*>(x + (size_t)(r0 + row) * DD + k0 + c4 * 4);
      v4bf bv = { (bf16_t)xv.x, (bf16_t)xv.y, (bf16_t)xv.z, (bf16_t)xv.w };
      *reinterpret_cast<v4bf*>(&Xs[row * 72 + c4 * 4]) = bv;
    }
    // Stage W tile transposed (convert f32 -> bf16)
#pragma unroll
    for (int i = 0; i < 4; ++i) {
      int idx = i * 256 + tid;               // 1024 float4 loads
      int d  = idx >> 4;
      int e4 = idx & 15;
      float4 wv = *reinterpret_cast<const float4*>(W + (size_t)(k0 + d) * DH + e4 * 4);
      Ws[(e4 * 4 + 0) * 72 + d] = (bf16_t)wv.x;
      Ws[(e4 * 4 + 1) * 72 + d] = (bf16_t)wv.y;
      Ws[(e4 * 4 + 2) * 72 + d] = (bf16_t)wv.z;
      Ws[(e4 * 4 + 3) * 72 + d] = (bf16_t)wv.w;
    }
    __syncthreads();
#pragma unroll
    for (int ks = 0; ks < 64; ks += 32) {
      v16bf a0 = load_frag(&Xs[0], wm * 32 +  0, ks, 72);
      v16bf a1 = load_frag(&Xs[0], wm * 32 + 16, ks, 72);
      v16bf b0 = load_frag(&Ws[0], wn * 32 +  0, ks, 72);
      v16bf b1 = load_frag(&Ws[0], wn * 32 + 16, ks, 72);
      c[0][0] = wmma_bf16(a0, b0, c[0][0]);
      c[0][1] = wmma_bf16(a0, b1, c[0][1]);
      c[1][0] = wmma_bf16(a1, b0, c[1][0]);
      c[1][1] = wmma_bf16(a1, b1, c[1][1]);
    }
    __syncthreads();
  }

  // Epilogue: tile never crosses a batch boundary (2048 % 128 == 0), and mat
  // is block-uniform -> one uniform branch, branch-free store loops inside.
  const int bidx = r0 >> 11;                 // batch index (uniform)
  const int t00  = (r0 & 2047) + wm * 32;    // base t for this wave
  const size_t bh = (size_t)(bidx * HH + h);

  if (mat == 0) {
    bf16_t* dst = Qb + bh * TT * DH;
#pragma unroll
    for (int mi = 0; mi < 2; ++mi)
#pragma unroll
      for (int ni = 0; ni < 2; ++ni) {
        int e = wn * 32 + ni * 16 + lanen;
#pragma unroll
        for (int r = 0; r < 8; ++r) {
          int t = t00 + mi * 16 + r + 8 * hf;
          dst[(size_t)t * DH + e] = (bf16_t)(c[mi][ni][r] * 0.125f);
        }
      }
  } else if (mat == 1) {
    bf16_t* dst = Kb + bh * TT * DH;
#pragma unroll
    for (int mi = 0; mi < 2; ++mi)
#pragma unroll
      for (int ni = 0; ni < 2; ++ni) {
        int e = wn * 32 + ni * 16 + lanen;
#pragma unroll
        for (int r = 0; r < 8; ++r) {
          int t = t00 + mi * 16 + r + 8 * hf;
          dst[(size_t)t * DH + e] = (bf16_t)c[mi][ni][r];
        }
      }
  } else {
    // V transposed [dh][t]: consecutive r -> consecutive t -> packed b32 stores
    bf16_t* dst = Vt + bh * DH * TT;
#pragma unroll
    for (int mi = 0; mi < 2; ++mi)
#pragma unroll
      for (int ni = 0; ni < 2; ++ni) {
        int e = wn * 32 + ni * 16 + lanen;
#pragma unroll
        for (int r = 0; r < 8; r += 2) {
          int t = t00 + mi * 16 + r + 8 * hf;   // even
          v2bf pv = { (bf16_t)c[mi][ni][r], (bf16_t)c[mi][ni][r + 1] };
          *reinterpret_cast<v2bf*>(&dst[(size_t)e * TT + t]) = pv;
        }
      }
  }
}

// ---------------------------------------------------------------------------
// Kernel 2: flash attention. grid (B*H=64, T/128=16). Each wave owns 16 query
// rows, streams keys in 64-wide tiles with online softmax; O accumulated in
// WMMA f32 accumulators. P goes C-layout -> LDS -> A-layout once per tile.
// ---------------------------------------------------------------------------
__global__ __launch_bounds__(256)
void mha_flash_attn_kernel(const bf16_t* __restrict__ Qb,
                           const bf16_t* __restrict__ Kb,
                           const bf16_t* __restrict__ Vt,
                           bf16_t* __restrict__ Ob) {
  __shared__ bf16_t Pl[8][16 * 72];          // per-wave 16x64 P tile (+pad)
  const int tid = threadIdx.x, lane = tid & 31, wid = tid >> 5;
  const int lanen = lane & 15, hf = lane >> 4;
  const int bh = blockIdx.x;                 // 0..63
  const int b = bh >> 4, h = bh & 15;
  const int q0 = blockIdx.y * 128 + wid * 16;
  const bf16_t* Qh = Qb + (size_t)bh * TT * DH;
  const bf16_t* Kh = Kb + (size_t)bh * TT * DH;
  const bf16_t* Vh = Vt + (size_t)bh * DH * TT;

  v16bf qf[2];
  qf[0] = load_frag(Qh, q0,  0, DH);
  qf[1] = load_frag(Qh, q0, 32, DH);

  v8f o[4] = {};
  float mrun[8], lrun[8];
#pragma unroll
  for (int r = 0; r < 8; ++r) { mrun[r] = -3.0e38f; lrun[r] = 0.0f; }

  for (int kt = 0; kt < TT; kt += 64) {
    if (kt + 64 < TT) {
      __builtin_prefetch(Kh + (size_t)(kt + 64) * DH, 0, 1);
      __builtin_prefetch(Vh + (size_t)(kt + 64), 0, 1);
    }
    // S = Q * K^T  (scale already folded into Q)
    v8f s[4] = {};
#pragma unroll
    for (int j = 0; j < 2; ++j) {
#pragma unroll
      for (int n = 0; n < 4; ++n) {
        v16bf bk = load_frag(Kh, kt + n * 16, j * 32, DH);
        s[n] = wmma_bf16(qf[j], bk, s[n]);
      }
    }
    // online softmax: row stats via xor-shuffle across the 16-lane half
    float alpha[8];
#pragma unroll
    for (int r = 0; r < 8; ++r) {
      float mx = fmaxf(fmaxf(s[0][r], s[1][r]), fmaxf(s[2][r], s[3][r]));
#pragma unroll
      for (int msk = 1; msk < 16; msk <<= 1) mx = fmaxf(mx, __shfl_xor(mx, msk, 32));
      float mn = fmaxf(mrun[r], mx);
      alpha[r] = __expf(mrun[r] - mn);
      mrun[r] = mn;
    }
    float rs[8];
#pragma unroll
    for (int r = 0; r < 8; ++r) rs[r] = 0.0f;
#pragma unroll
    for (int n = 0; n < 4; ++n)
#pragma unroll
      for (int r = 0; r < 8; ++r) {
        float p = __expf(s[n][r] - mrun[r]);
        s[n][r] = p;
        rs[r] += p;
      }
#pragma unroll
    for (int r = 0; r < 8; ++r) {
#pragma unroll
      for (int msk = 1; msk < 16; msk <<= 1) rs[r] += __shfl_xor(rs[r], msk, 32);
      lrun[r] = lrun[r] * alpha[r] + rs[r];
    }
#pragma unroll
    for (int n = 0; n < 4; ++n)
#pragma unroll
      for (int r = 0; r < 8; ++r) o[n][r] *= alpha[r];

    // C-layout -> row-major bf16 P in LDS
    bf16_t* P = &Pl[wid][0];
#pragma unroll
    for (int n = 0; n < 4; ++n)
#pragma unroll
      for (int r = 0; r < 8; ++r)
        P[(r + 8 * hf) * 72 + n * 16 + lanen] = (bf16_t)s[n][r];
    __syncthreads();

    // O += P * V   (V^T is row-major [dh][t] -> direct B-fragments)
    v16bf pa0 = load_frag(P, 0,  0, 72);
    v16bf pa1 = load_frag(P, 0, 32, 72);
#pragma unroll
    for (int n = 0; n < 4; ++n) {
      v16bf bv0 = load_frag(Vh, n * 16, kt +  0, TT);
      v16bf bv1 = load_frag(Vh, n * 16, kt + 32, TT);
      o[n] = wmma_bf16(pa0, bv0, o[n]);
      o[n] = wmma_bf16(pa1, bv1, o[n]);
    }
    __syncthreads();
  }

  // normalize and store concat-head layout [b, t, h*64 + e]
#pragma unroll
  for (int n = 0; n < 4; ++n)
#pragma unroll
    for (int r = 0; r < 8; ++r) {
      int t = q0 + r + 8 * hf;
      float val = o[n][r] / lrun[r];
      Ob[((size_t)(b * TT + t) * (HH * DH)) + h * DH + n * 16 + lanen] = (bf16_t)val;
    }
}

// ---------------------------------------------------------------------------
// Kernel 3: output projection + bias.  Ob[8192,1024]bf16 x Wo[1024,1024]f32
// grid (64, 16): 128-row x 64-col tiles. A-fragments stream straight from
// global (bf16 row-major); Wo transposed+converted through LDS.
// ---------------------------------------------------------------------------
__global__ __launch_bounds__(256)
void mha_out_proj_kernel(const bf16_t* __restrict__ Ob,
                         const float* __restrict__ Wo,
                         const float* __restrict__ bo,
                         float* __restrict__ out) {
  __shared__ bf16_t Ws[64 * 72];
  const int tid = threadIdx.x, lane = tid & 31, wid = tid >> 5;
  const int lanen = lane & 15, hf = lane >> 4;
  const int wm = wid >> 1, wn = wid & 1;
  const int r0 = blockIdx.x * 128;
  const int n0 = blockIdx.y * 64;

  v8f c[2][2] = {};

  for (int k0 = 0; k0 < DD; k0 += 64) {
#pragma unroll
    for (int i = 0; i < 4; ++i) {
      int idx = i * 256 + tid;
      int d = idx >> 4, e4 = idx & 15;
      float4 wv = *reinterpret_cast<const float4*>(Wo + (size_t)(k0 + d) * DD + n0 + e4 * 4);
      Ws[(e4 * 4 + 0) * 72 + d] = (bf16_t)wv.x;
      Ws[(e4 * 4 + 1) * 72 + d] = (bf16_t)wv.y;
      Ws[(e4 * 4 + 2) * 72 + d] = (bf16_t)wv.z;
      Ws[(e4 * 4 + 3) * 72 + d] = (bf16_t)wv.w;
    }
    __syncthreads();
#pragma unroll
    for (int ks = 0; ks < 64; ks += 32) {
      v16bf a0 = load_frag(Ob, r0 + wm * 32 +  0, k0 + ks, DD);
      v16bf a1 = load_frag(Ob, r0 + wm * 32 + 16, k0 + ks, DD);
      v16bf b0 = load_frag(&Ws[0], wn * 32 +  0, ks, 72);
      v16bf b1 = load_frag(&Ws[0], wn * 32 + 16, ks, 72);
      c[0][0] = wmma_bf16(a0, b0, c[0][0]);
      c[0][1] = wmma_bf16(a0, b1, c[0][1]);
      c[1][0] = wmma_bf16(a1, b0, c[1][0]);
      c[1][1] = wmma_bf16(a1, b1, c[1][1]);
    }
    __syncthreads();
  }

#pragma unroll
  for (int mi = 0; mi < 2; ++mi)
#pragma unroll
    for (int ni = 0; ni < 2; ++ni) {
      int col = n0 + wn * 32 + ni * 16 + lanen;
      float bias = bo[col];
#pragma unroll
      for (int r = 0; r < 8; ++r) {
        int row = r0 + wm * 32 + mi * 16 + r + 8 * hf;
        out[(size_t)row * DD + col] = c[mi][ni][r] + bias;
      }
    }
}

extern "C" void kernel_launch(void* const* d_in, const int* in_sizes, int n_in,
                              void* d_out, int out_size, void* d_ws, size_t ws_size,
                              hipStream_t stream) {
  const float* x  = (const float*)d_in[0];
  const float* Wq = (const float*)d_in[1];
  const float* Wk = (const float*)d_in[2];
  const float* Wv = (const float*)d_in[3];
  const float* Wo = (const float*)d_in[4];
  const float* bo = (const float*)d_in[5];
  float* out = (float*)d_out;

  const size_t elems = (size_t)BB * HH * TT * DH;   // 8,388,608 per buffer
  bf16_t* Qb = (bf16_t*)d_ws;
  bf16_t* Kb = Qb + elems;
  bf16_t* Vt = Kb + elems;
  bf16_t* Ob = Vt + elems;

  mha_qkv_proj_kernel<<<dim3(64, 48), 256, 0, stream>>>(x, Wq, Wk, Wv, Qb, Kb, Vt);
  mha_flash_attn_kernel<<<dim3(64, 16), 256, 0, stream>>>(Qb, Kb, Vt, Ob);
  mha_out_proj_kernel<<<dim3(64, 16), 256, 0, stream>>>(Ob, Wo, bo, out);
}
